// RelationalEncoder_38646115729912
// MI455X (gfx1250) — matrix operand order repaired and checked by reading, not compiled, not run on hardware
//
#include <hip/hip_runtime.h>
#include <hip/hip_bf16.h>
#include <math.h>

// ---------------- CDNA5 WMMA types ----------------
typedef __bf16 v16bf __attribute__((ext_vector_type(16)));
typedef float  v8f   __attribute__((ext_vector_type(8)));

union ABFrag { v16bf v; unsigned int w[8]; uint4 q[2]; };

#define WAVES_PER_WG 8
#define BLOCK        256
#define GRID         1024
#define ASTR         136            // per-wave LDS activation tile stride (bf16 elems); 272B rows, 16B-aligned
#define NTILES       16384          // 262144 pair-rows / 16
// problem constants: B=2,T=32,N=64 -> rows = 2*32*64*64 = 262144, out dim 256

__device__ __forceinline__ unsigned short f2bf(float f) {
  unsigned int x = __float_as_uint(f);
  x += 0x7FFFu + ((x >> 16) & 1u);        // round-to-nearest-even
  return (unsigned short)(x >> 16);
}
__device__ __forceinline__ unsigned int f2bf_pk(float lo, float hi) {
  return (unsigned int)f2bf(lo) | ((unsigned int)f2bf(hi) << 16);
}

// A fragment (16x32 bf16): lane l -> row m=l&15, half h=l>>4.
// VGPR e holds K = k0 + (e/4)*16 + h*8 + (e%4)*2 + {0,1}   (ISA 7.12.2)
// => e=0..3 are the 8 contiguous ushorts at k0+h*8, e=4..7 at k0+16+h*8: two b128 loads.
__device__ __forceinline__ v16bf load_a(const unsigned short* abuf, int lane, int k0) {
  const int m = lane & 15, h = lane >> 4;
  const unsigned short* row = abuf + m * ASTR + k0 + h * 8;
  ABFrag f;
  f.q[0] = *(const uint4*)(row);
  f.q[1] = *(const uint4*)(row + 16);
  return f.v;
}

// B fragment (32x16 bf16) from transposed weight wT[d][n] (row-major, stride nstr):
// lane l -> N = n0 + (l&15), h = l>>4 ; element e -> K = d0 + (e/8)*16 + h*8 + e%8 (same as A)
// strided along K: 16 u16 loads packed in pairs.
__device__ __forceinline__ v16bf load_b(const unsigned short* wT, int lane, int d0, int n0, int nstr) {
  const unsigned short* p = wT + (d0 + (lane >> 4) * 8) * nstr + n0 + (lane & 15);
  ABFrag f;
#pragma unroll
  for (int i = 0; i < 4; ++i) {
    f.w[i]     = (unsigned int)p[(2 * i) * nstr] | ((unsigned int)p[(2 * i + 1) * nstr] << 16);
    f.w[4 + i] = (unsigned int)p[(16 + 2 * i) * nstr] | ((unsigned int)p[(17 + 2 * i) * nstr] << 16);
  }
  return f.v;
}

#define WMMA_BF16(a, b, c) \
  __builtin_amdgcn_wmma_f32_16x16x32_bf16(false, (a), false, (b), (short)0, (c), false, false)

__global__ __launch_bounds__(BLOCK) void rel_enc_fused(
    const float* __restrict__ pos, const float* __restrict__ vel,
    const float* __restrict__ att, const float* __restrict__ w1,
    const float* __restrict__ b1,  const float* __restrict__ g,
    const float* __restrict__ w2,  const float* __restrict__ b2,
    const float* __restrict__ wa,  const float* __restrict__ ba,
    float* __restrict__ out)
{
  extern __shared__ unsigned short smem[];
  unsigned short* w1T   = smem;                 // [64][128]  bf16  (w1^T)
  unsigned short* w2T   = w1T + 64 * 128;       // [128][128] bf16  (w2^T)
  unsigned short* waT   = w2T + 128 * 128;      // [128][256] bf16  (wa^T)
  unsigned short* abase = waT + 128 * 256;      // per-wave 16 x ASTR activation tiles

  const int tid  = threadIdx.x;
  const int lane = tid & 31;
  const int wave = tid >> 5;
  unsigned short* abuf = abase + wave * (16 * ASTR);

  // ---- stage weights: f32 global -> bf16 transposed LDS, packed b32 writes ----
  // w1[k][d] -> w1T[d][k]; each thread packs (k,k+1) for one d.
  for (int p = tid; p < (128 / 2) * 64; p += BLOCK) {
    const int d = p & 63, k = (p >> 6) << 1;
    ((unsigned int*)w1T)[(d * 128 + k) >> 1] = f2bf_pk(w1[k * 64 + d], w1[(k + 1) * 64 + d]);
  }
  for (int p = tid; p < (128 / 2) * 128; p += BLOCK) {   // w2[k][d] -> w2T[d][k]
    const int d = p & 127, k = (p >> 7) << 1;
    ((unsigned int*)w2T)[(d * 128 + k) >> 1] = f2bf_pk(w2[k * 128 + d], w2[(k + 1) * 128 + d]);
  }
  for (int p = tid; p < (256 / 2) * 128; p += BLOCK) {   // wa[n][d] -> waT[d][n]
    const int d = p & 127, n = (p >> 7) << 1;
    ((unsigned int*)waT)[(d * 256 + n) >> 1] = f2bf_pk(wa[n * 128 + d], wa[(n + 1) * 128 + d]);
  }
  __syncthreads();

  // per-lane column-resident vectors (lane = output column within 16-wide tile)
  const int col = lane & 15;
  const int h   = lane >> 4;
  float gcol[8], b1col[8], b2col[8], bacol[16];
#pragma unroll
  for (int nt = 0; nt < 8; ++nt) {
    gcol[nt]  = g[nt * 16 + col];
    b1col[nt] = b1[nt * 16 + col];
    b2col[nt] = b2[nt * 16 + col];
  }
#pragma unroll
  for (int nt = 0; nt < 16; ++nt) bacol[nt] = ba[nt * 16 + col];

  const float2* pos2 = (const float2*)pos;
  const float2* vel2 = (const float2*)vel;
  const float2* att2 = (const float2*)att;

  const int waveGlobal = blockIdx.x * WAVES_PER_WG + wave;
  const int totalWaves = gridDim.x * WAVES_PER_WG;

  for (int tile = waveGlobal; tile < NTILES; tile += totalWaves) {
    // ================= features: 16 pair-rows, one per lane 0..15 =============
    if (lane < 16) {
      const int r  = tile * 16 + lane;
      const int j  = r & 63;
      const int i  = (r >> 6) & 63;
      const int bt = r >> 12;                 // b*32 + t
      const float2 pi = pos2[bt * 64 + i], pj = pos2[bt * 64 + j];
      const float2 vi = vel2[bt * 64 + i], vj = vel2[bt * 64 + j];
      const float2 ai = att2[bt * 64 + i], aj = att2[bt * 64 + j];

      float f[64];
      float dx = pi.x - pj.x; dx -= rintf(dx * (1.0f / 1024.0f)) * 1024.0f;
      float dy = pi.y - pj.y; dy -= rintf(dy * (1.0f / 1024.0f)) * 1024.0f;
      const float dvx = vi.x - vj.x, dvy = vi.y - vj.y;
      const float dist = sqrtf(dx * dx + dy * dy) + 1e-6f;
      const float ird  = 1.0f / (dist + 1e-6f);
      const float dirx = dx * ird, diry = dy * ird;
      const float closing = (dvx * dx + dvy * dy) * ird;
      f[0] = dx;  f[1] = dy;  f[2] = dvx; f[3] = dvy;
      f[4] = dist;
      f[5] = 1.0f / (dist + 0.1f);
      f[6] = sqrtf(dvx * dvx + dvy * dvy);
      f[7] = closing;
      f[8] = dirx; f[9] = diry;
      f[10] = logf(dist + 1.0f);
      f[11] = dist / fmaxf(closing, 0.001f);
      f[12] = dirx * ai.x + diry * ai.y;
      f[13] = dirx * ai.y - diry * ai.x;
      f[14] = -(dirx * aj.x + diry * aj.y);
      f[15] = -dirx * aj.y + diry * aj.x;
      f[16] = ai.x * aj.x + ai.y * aj.y;
      f[17] = ai.x * aj.y - ai.y * aj.x;
      const float sc = 6.28318530717958647692f / 1024.0f;
      const float sx = dx * sc, sy = dy * sc;
#pragma unroll
      for (int band = 0; band < 8; ++band) {
        const float fm = (float)(1 << band);
        float ssx, csx, ssy, csy;
        __sincosf(sx * fm, &ssx, &csx);
        __sincosf(sy * fm, &ssy, &csy);
        f[18 + band * 4 + 0] = ssx;
        f[18 + band * 4 + 1] = ssy;
        f[18 + band * 4 + 2] = csx;
        f[18 + band * 4 + 3] = csy;
      }
#pragma unroll
      for (int k = 50; k < 64; ++k) f[k] = 0.0f;   // pad to 64

      // pack to bf16 pairs and store as 8 x b128
      unsigned int fu[32];
#pragma unroll
      for (int q2 = 0; q2 < 32; ++q2) fu[q2] = f2bf_pk(f[2 * q2], f[2 * q2 + 1]);
      uint4* arow4 = (uint4*)(abuf + lane * ASTR);
#pragma unroll
      for (int q4 = 0; q4 < 8; ++q4) arow4[q4] = *(const uint4*)&fu[q4 * 4];
    }
    asm volatile("" ::: "memory");   // LDS ops are in-order per wave; just stop reordering

    // ================= GEMM1: h = raw(16x64) @ w1T(64x128) ====================
    const v16bf a0 = load_a(abuf, lane, 0);
    const v16bf a1 = load_a(abuf, lane, 32);
    v8f hfr[8];
#pragma unroll
    for (int nt = 0; nt < 8; ++nt) {
      v8f c = {};
      c = WMMA_BF16(a0, load_b(w1T, lane, 0,  nt * 16, 128), c);
      c = WMMA_BF16(a1, load_b(w1T, lane, 32, nt * 16, 128), c);
      hfr[nt] = c;
    }
    // + b1, row-wise sum of squares (row = v + 8*h; cols spread over 16 lanes of a half)
    float ss[8];
#pragma unroll
    for (int v = 0; v < 8; ++v) ss[v] = 0.0f;
#pragma unroll
    for (int nt = 0; nt < 8; ++nt)
#pragma unroll
      for (int v = 0; v < 8; ++v) {
        const float x = hfr[nt][v] + b1col[nt];
        hfr[nt][v] = x;
        ss[v] += x * x;
      }
#pragma unroll
    for (int m = 1; m < 16; m <<= 1)       // xor masks 1..8 stay within each 16-lane half
#pragma unroll
      for (int v = 0; v < 8; ++v) ss[v] += __shfl_xor(ss[v], m, 32);
    float rs[8];
#pragma unroll
    for (int v = 0; v < 8; ++v) rs[v] = rsqrtf(ss[v] * (1.0f / 128.0f) + 1e-5f);

    // silu(rmsnorm * g) -> bf16 back into abuf (C-layout -> row-major via LDS)
    asm volatile("" ::: "memory");
#pragma unroll
    for (int nt = 0; nt < 8; ++nt)
#pragma unroll
      for (int v = 0; v < 8; ++v) {
        float x = hfr[nt][v] * rs[v] * gcol[nt];
        x = x / (1.0f + __expf(-x));
        abuf[(v + 8 * h) * ASTR + nt * 16 + col] = f2bf(x);
      }
    asm volatile("" ::: "memory");

    // ================= GEMM2: trunk = h(16x128) @ w2T(128x128) + b2 ===========
    v16bf ah[4];
#pragma unroll
    for (int kt = 0; kt < 4; ++kt) ah[kt] = load_a(abuf, lane, kt * 32);
#pragma unroll
    for (int nt = 0; nt < 8; ++nt) {
      v8f c = {};
#pragma unroll
      for (int kt = 0; kt < 4; ++kt)
        c = WMMA_BF16(ah[kt], load_b(w2T, lane, kt * 32, nt * 16, 128), c);
      hfr[nt] = c;
    }
    asm volatile("" ::: "memory");
#pragma unroll
    for (int nt = 0; nt < 8; ++nt)
#pragma unroll
      for (int v = 0; v < 8; ++v)
        abuf[(v + 8 * h) * ASTR + nt * 16 + col] = f2bf(hfr[nt][v] + b2col[nt]);
    asm volatile("" ::: "memory");

    // ================= GEMM3: out = trunk(16x128) @ waT(128x256) + ba =========
#pragma unroll
    for (int kt = 0; kt < 4; ++kt) ah[kt] = load_a(abuf, lane, kt * 32);
    const long rowg0 = (long)tile * 16;
#pragma unroll
    for (int nt = 0; nt < 16; ++nt) {
      v8f c = {};
#pragma unroll
      for (int kt = 0; kt < 4; ++kt)
        c = WMMA_BF16(ah[kt], load_b(waT, lane, kt * 32, nt * 16, 256), c);
      float* op = out + (rowg0 + 8 * h) * 256 + nt * 16 + col;
#pragma unroll
      for (int v = 0; v < 8; ++v)
        op[(long)v * 256] = c[v] + bacol[nt];
    }
  }
}

extern "C" void kernel_launch(void* const* d_in, const int* in_sizes, int n_in,
                              void* d_out, int out_size, void* d_ws, size_t ws_size,
                              hipStream_t stream) {
  (void)in_sizes; (void)n_in; (void)out_size; (void)d_ws; (void)ws_size;
  const float* pos = (const float*)d_in[0];
  const float* vel = (const float*)d_in[1];
  const float* att = (const float*)d_in[2];
  const float* w1  = (const float*)d_in[3];
  const float* b1  = (const float*)d_in[4];
  const float* g   = (const float*)d_in[5];
  const float* w2  = (const float*)d_in[6];
  const float* b2  = (const float*)d_in[7];
  const float* wa  = (const float*)d_in[8];
  const float* ba  = (const float*)d_in[9];
  // 112 KB weights + 8 waves * 4.25 KB activation tiles = ~146 KB dynamic LDS
  // (CDNA5 WGP allows up to 320 KB per workgroup)
  const size_t shmem =
      (size_t)(64 * 128 + 128 * 128 + 128 * 256 + WAVES_PER_WG * 16 * ASTR) *
      sizeof(unsigned short);
  rel_enc_fused<<<GRID, BLOCK, shmem, stream>>>(pos, vel, att, w1, b1, g, w2, b2,
                                                wa, ba, (float*)d_out);
}
